// WindowAttention_12128987644464
// MI455X (gfx1250) — matrix operand order repaired and checked by reading, not compiled, not run on hardware
//
#include <hip/hip_runtime.h>

typedef __bf16 bf16_t;
typedef __attribute__((ext_vector_type(16))) __bf16 v16bf;
typedef __attribute__((ext_vector_type(8)))  __bf16 v8bf;
typedef __attribute__((ext_vector_type(4)))  __bf16 v4bf;
typedef __attribute__((ext_vector_type(8)))  float  v8f;
typedef __attribute__((ext_vector_type(4)))  float  v4f;

// ---------------------------------------------------------------------------
// WMMA fragment helpers (CDNA5 wave32 layouts, CDNA5 ISA §7.12.2)
// ---------------------------------------------------------------------------

// A-matrix 16x32 bf16 fragment from a row-major bf16 matrix.
// lane L: row = L&15.  lanes 0-15 hold K 0..15;
// lanes 16-31 hold K 16..31.
__device__ __forceinline__ v16bf load_a_frag(const bf16_t* p0, int stride, int lane) {
    const bf16_t* p = p0 + (lane & 15) * stride + ((lane & 16) ? 16 : 0);
    v8bf lo = *(const v8bf*)(p);
    v8bf hi = *(const v8bf*)(p + 8);
    v16bf r;
#pragma unroll
    for (int e = 0; e < 8; ++e) { r[e] = lo[e]; r[e + 8] = hi[e]; }
    return r;
}

// B-matrix 32x16 bf16 fragment. B[k][n] = src[n*stride + k] (WMMA column n
// indexes rows of a row-major array, K contiguous).
// lanes 0-15 hold K 0..15; lanes 16-31 hold K 16..31.
__device__ __forceinline__ v16bf load_b_frag(const bf16_t* p0, int stride, int lane) {
    const bf16_t* p = p0 + (lane & 15) * stride + ((lane & 16) ? 16 : 0);
    v8bf lo = *(const v8bf*)(p);
    v8bf hi = *(const v8bf*)(p + 8);
    v16bf r;
#pragma unroll
    for (int e = 0; e < 8; ++e) { r[e] = lo[e]; r[e + 8] = hi[e]; }
    return r;
}

// B fragment from a row-major bf16 matrix in GLOBAL memory (pre-converted
// weights in d_ws): 32 contiguous bytes per lane -> 2x global_load_b128.
__device__ __forceinline__ v16bf load_b_frag_gbf(const bf16_t* p0, int stride, int lane) {
    const bf16_t* p = p0 + (size_t)(lane & 15) * stride + ((lane & 16) ? 16 : 0);
    v8bf lo = *(const v8bf*)(p);
    v8bf hi = *(const v8bf*)(p + 8);
    v16bf r;
#pragma unroll
    for (int e = 0; e < 8; ++e) { r[e] = lo[e]; r[e + 8] = hi[e]; }
    return r;
}

// Fallback: B fragment from a row-major fp32 matrix in global memory.
__device__ __forceinline__ v16bf load_b_frag_gf32(const float* p0, int stride, int lane) {
    const float* p = p0 + (size_t)(lane & 15) * stride + ((lane & 16) ? 16 : 0);
    v4f a = *(const v4f*)(p);
    v4f b = *(const v4f*)(p + 4);
    v4f c = *(const v4f*)(p + 8);
    v4f d = *(const v4f*)(p + 12);
    v16bf r;
#pragma unroll
    for (int e = 0; e < 4; ++e) {
        r[e]      = (__bf16)a[e];
        r[e + 4]  = (__bf16)b[e];
        r[e + 8]  = (__bf16)c[e];
        r[e + 12] = (__bf16)d[e];
    }
    return r;
}

__device__ __forceinline__ v8f wmma_bf16(v16bf a, v16bf b, v8f c) {
    return __builtin_amdgcn_wmma_f32_16x16x32_bf16(
        /*neg_a=*/false, a, /*neg_b=*/false, b,
        /*c_mod=*/(short)0, c, /*reuse_a=*/false, /*reuse_b=*/false);
}

// ---------------------------------------------------------------------------
// Prologue: fp32 weights -> bf16 scratch (w_qkv: 196608 elems, w_proj: 65536)
// ---------------------------------------------------------------------------
__global__ __launch_bounds__(256)
void convert_weights(const float* __restrict__ wq, const float* __restrict__ wp,
                     bf16_t* __restrict__ out) {
    const int i4 = blockIdx.x * blockDim.x + threadIdx.x;   // float4 index, 0..65535
    v4f v;
    if (i4 < 49152) v = *(const v4f*)(wq + (size_t)i4 * 4);
    else            v = *(const v4f*)(wp + (size_t)(i4 - 49152) * 4);
    v4bf o4;
#pragma unroll
    for (int e = 0; e < 4; ++e) o4[e] = (__bf16)v[e];
    *(v4bf*)(out + (size_t)i4 * 4) = o4;
}

// ---------------------------------------------------------------------------
// Fused window attention: qkv GEMM -> per-head attention+softmax -> proj GEMM
// One workgroup (8 waves) per window.  LDS (dynamic, 160 KB => 2 WGs/WGP):
//   elems [0,     16384)  x tile (bf16 64x256)  -- overlaid by P [H][16][64]
//   elems [16384, 32768)  q  [H][64][32] (pre-scaled)
//   elems [32768, 49152)  k  [H][64][32]
//   elems [49152, 65536)  vT [H][32][64]
//   elems [65536, 81920)  o  [64][256] attention output
// ---------------------------------------------------------------------------
template <bool PRECONV>
__global__ __launch_bounds__(256)
void win_attn_fused(const float* __restrict__ x,
                    const float* __restrict__ w_qkv_f,
                    const bf16_t* __restrict__ w_qkv_bf,
                    const float* __restrict__ b_qkv,
                    const float* __restrict__ bias_table,
                    const float* __restrict__ w_proj_f,
                    const bf16_t* __restrict__ w_proj_bf,
                    const float* __restrict__ b_proj,
                    float* __restrict__ out) {
    constexpr int N = 64, C = 256, H = 8;
    constexpr float SCALE = 0.17677669529663687f;  // 32^-0.5

    extern __shared__ char smem_raw[];
    bf16_t* base   = (bf16_t*)smem_raw;
    bf16_t* x_lds  = base;            // 64*256
    bf16_t* p_lds  = base;            // [H][16][64] strip buffer (reuses x region)
    bf16_t* q_lds  = base + 16384;    // [H][64][32]
    bf16_t* k_lds  = base + 32768;    // [H][64][32]
    bf16_t* vT_lds = base + 49152;    // [H][32][64]
    bf16_t* o_lds  = base + 65536;    // [64][256]

    const int b      = blockIdx.x;
    const int tid    = threadIdx.x;
    const int lane   = tid & 31;
    const int wave   = tid >> 5;
    const int n_lane = lane & 15;
    const int rhalf  = (lane & 16) ? 8 : 0;

    // ---- Stage 1: x[b] (fp32) -> LDS bf16 ---------------------------------
    const float* xg = x + (size_t)b * (N * C);
#pragma unroll
    for (int it = 0; it < 16; ++it) {
        int i4 = tid + it * 256;              // float4 index within 64x256
        v4f v = *(const v4f*)(xg + (size_t)i4 * 4);
        v4bf w;
#pragma unroll
        for (int e = 0; e < 4; ++e) w[e] = (__bf16)v[e];
        *(v4bf*)(x_lds + i4 * 4) = w;
    }
    __syncthreads();

    // ---- Stage 2: qkv = x @ w_qkv^T + b_qkv, routed into q/k/vT -----------
    const int i_strip = wave & 3;   // 16-row strip of tokens
    const int cgrp    = wave >> 2;  // column half
    {
        v16bf a_x[8];
#pragma unroll
        for (int kk = 0; kk < 8; ++kk)
            a_x[kk] = load_a_frag(x_lds + (i_strip * 16) * C + kk * 32, C, lane);

        const int jt_end = cgrp * 24 + 24;
        for (int jt = cgrp * 24; jt < jt_end; ++jt) {
            const int j0 = jt * 16;
            // prefetch next tile's weight block (global_prefetch_b8)
            if (jt + 1 < jt_end) {
                if (PRECONV)
                    __builtin_prefetch(w_qkv_bf + (size_t)(j0 + 16 + n_lane) * C, 0, 3);
                else
                    __builtin_prefetch(w_qkv_f + (size_t)(j0 + 16 + n_lane) * C, 0, 3);
            }
            v8f acc = {};
#pragma unroll
            for (int kk = 0; kk < 8; ++kk) {
                v16bf bf = PRECONV
                    ? load_b_frag_gbf(w_qkv_bf + (size_t)j0 * C + kk * 32, C, lane)
                    : load_b_frag_gf32(w_qkv_f + (size_t)j0 * C + kk * 32, C, lane);
                acc = wmma_bf16(a_x[kk], bf, acc);
            }
            // --- route to q / k / vT: segment decision is WAVE-UNIFORM -----
            const int   seg  = j0 >> 8;            // 0=q, 1=k, 2=v (scalar)
            const int   hU   = (j0 >> 5) & 7;      // head (scalar)
            const int   dd   = (j0 & 31) + n_lane; // 0..31 (per-lane)
            const int   row0 = i_strip * 16 + rhalf;
            const float bias = b_qkv[j0 + n_lane];
            if (seg == 0) {                        // Q (pre-scaled)
                bf16_t* qp = q_lds + (hU * 64 + row0) * 32 + dd;
#pragma unroll
                for (int r = 0; r < 8; ++r)
                    qp[r * 32] = (__bf16)((acc[r] + bias) * SCALE);
            } else if (seg == 1) {                 // K
                bf16_t* kp = k_lds + (hU * 64 + row0) * 32 + dd;
#pragma unroll
                for (int r = 0; r < 8; ++r)
                    kp[r * 32] = (__bf16)(acc[r] + bias);
            } else {                               // V transposed: rows contiguous
                v8bf pk;
#pragma unroll
                for (int r = 0; r < 8; ++r) pk[r] = (__bf16)(acc[r] + bias);
                *(v8bf*)(vT_lds + (hU * 32 + dd) * 64 + row0) = pk;
            }
        }
    }
    __syncthreads();

    // ---- Stage 3: per-head attention (wave w == head w) -------------------
    {
        const int h = wave;
        const bf16_t* qh = q_lds + h * 64 * 32;
        const bf16_t* kh = k_lds + h * 64 * 32;
        const bf16_t* vh = vT_lds + h * 32 * 64;
        bf16_t*       ph = p_lds + h * 16 * 64;    // per-head strip buffer

        v16bf kb[4];                               // K^T tiles (K=d=32, one WMMA each)
#pragma unroll
        for (int j = 0; j < 4; ++j) kb[j] = load_b_frag(kh + (j * 16) * 32, 32, lane);
        v16bf vb[2][2];                            // V tiles [kblock][dd-tile]
#pragma unroll
        for (int kb2 = 0; kb2 < 2; ++kb2)
#pragma unroll
            for (int dt = 0; dt < 2; ++dt)
                vb[kb2][dt] = load_b_frag(vh + dt * 16 * 64 + kb2 * 32, 64, lane);

        for (int i2 = 0; i2 < 4; ++i2) {           // query row strips
            v16bf qa = load_a_frag(qh + i2 * 16 * 32, 32, lane);
            v8f s[4];
#pragma unroll
            for (int j = 0; j < 4; ++j) { v8f z = {}; s[j] = wmma_bf16(qa, kb[j], z); }

            // bias + softmax over the 64 keys (row lives in one 16-lane half)
#pragma unroll
            for (int r = 0; r < 8; ++r) {
                const int lr = r + rhalf;                  // strip-local row
                const int nG = i2 * 16 + lr;               // query token
                float mx = -3.4028235e38f;
#pragma unroll
                for (int j = 0; j < 4; ++j) {
                    const int mG = j * 16 + n_lane;        // key token
                    const int dy = (nG >> 3) - (mG >> 3) + 7;
                    const int dx = (nG & 7) - (mG & 7) + 7;
                    float sv = s[j][r] + bias_table[(dy * 15 + dx) * H + h];
                    s[j][r] = sv;
                    mx = fmaxf(mx, sv);
                }
#pragma unroll
                for (int off = 1; off < 16; off <<= 1)
                    mx = fmaxf(mx, __shfl_xor(mx, off, 32));
                float sum = 0.f;
#pragma unroll
                for (int j = 0; j < 4; ++j) {
                    float e = __expf(s[j][r] - mx);
                    s[j][r] = e;
                    sum += e;
                }
#pragma unroll
                for (int off = 1; off < 16; off <<= 1)
                    sum += __shfl_xor(sum, off, 32);
                const float inv = 1.0f / sum;
#pragma unroll
                for (int j = 0; j < 4; ++j)
                    ph[lr * 64 + j * 16 + n_lane] = (__bf16)(s[j][r] * inv);
            }

            // O_strip = P_strip (16x64) @ V (64x32); LDS round-trip converts
            // the D-layout P into A-layout fragments.
            v8f o0 = {}, o1 = {};
#pragma unroll
            for (int kb2 = 0; kb2 < 2; ++kb2) {
                v16bf pa = load_a_frag(ph + kb2 * 32, 64, lane);
                o0 = wmma_bf16(pa, vb[kb2][0], o0);
                o1 = wmma_bf16(pa, vb[kb2][1], o1);
            }
#pragma unroll
            for (int r = 0; r < 8; ++r) {
                const int nG = i2 * 16 + r + rhalf;
                o_lds[nG * 256 + h * 32 + n_lane]      = (__bf16)o0[r];
                o_lds[nG * 256 + h * 32 + 16 + n_lane] = (__bf16)o1[r];
            }
        }
    }
    __syncthreads();

    // ---- Stage 4: out = O @ w_proj^T + b_proj -----------------------------
    {
        v16bf a_o[8];
#pragma unroll
        for (int kk = 0; kk < 8; ++kk)
            a_o[kk] = load_a_frag(o_lds + (i_strip * 16) * C + kk * 32, C, lane);

        float* outg = out + (size_t)b * (N * C);
        const int ct_end = cgrp * 8 + 8;
        for (int ct = cgrp * 8; ct < ct_end; ++ct) {
            const int n0 = ct * 16;
            if (ct + 1 < ct_end) {
                if (PRECONV)
                    __builtin_prefetch(w_proj_bf + (size_t)(n0 + 16 + n_lane) * C, 0, 3);
                else
                    __builtin_prefetch(w_proj_f + (size_t)(n0 + 16 + n_lane) * C, 0, 3);
            }
            v8f acc = {};
#pragma unroll
            for (int kk = 0; kk < 8; ++kk) {
                v16bf bf = PRECONV
                    ? load_b_frag_gbf(w_proj_bf + (size_t)n0 * C + kk * 32, C, lane)
                    : load_b_frag_gf32(w_proj_f + (size_t)n0 * C + kk * 32, C, lane);
                acc = wmma_bf16(a_o[kk], bf, acc);
            }
            const float bb = b_proj[n0 + n_lane];
#pragma unroll
            for (int r = 0; r < 8; ++r) {
                const int row = i_strip * 16 + r + rhalf;
                outg[(size_t)row * 256 + n0 + n_lane] = acc[r] + bb;
            }
        }
    }
}

extern "C" void kernel_launch(void* const* d_in, const int* in_sizes, int n_in,
                              void* d_out, int out_size, void* d_ws, size_t ws_size,
                              hipStream_t stream) {
    const float* x          = (const float*)d_in[0];
    const float* w_qkv      = (const float*)d_in[1];
    const float* b_qkv      = (const float*)d_in[2];
    const float* bias_table = (const float*)d_in[3];
    const float* w_proj     = (const float*)d_in[4];
    const float* b_proj     = (const float*)d_in[5];
    float*       out        = (float*)d_out;

    const int    B     = in_sizes[0] / (64 * 256);   // windows
    const size_t shmem = 81920 * 2;                  // 160 KB dynamic LDS

    const bool preconv = ws_size >= 262144u * sizeof(unsigned short);
    if (preconv) {
        bf16_t* w_bf = (bf16_t*)d_ws;                // [0,196608) qkv, [196608,262144) proj
        convert_weights<<<dim3(256), dim3(256), 0, stream>>>(w_qkv, w_proj, w_bf);
        win_attn_fused<true><<<dim3(B), dim3(256), shmem, stream>>>(
            x, w_qkv, w_bf, b_qkv, bias_table, w_proj, w_bf + 196608, b_proj, out);
    } else {
        win_attn_fused<false><<<dim3(B), dim3(256), shmem, stream>>>(
            x, w_qkv, nullptr, b_qkv, bias_table, w_proj, nullptr, b_proj, out);
    }
}
